// SimplePointNetCPL_86912958202267
// MI455X (gfx1250) — matrix operand order repaired and checked by reading, not compiled, not run on hardware
//
#include <hip/hip_runtime.h>
#include <hip/hip_bf16.h>

// ---------------------------------------------------------------------------
// Types for CDNA5 WMMA (wave32, 16x16x32 bf16 -> f32)
// ---------------------------------------------------------------------------
typedef __attribute__((ext_vector_type(16))) __bf16 v16bf;
typedef __attribute__((ext_vector_type(8)))  float  v8f;

union FragAB { v16bf v; unsigned int u32[8]; unsigned short u16[16]; };
union FragC  { v8f   v; float f[8]; };

__device__ __forceinline__ unsigned short f32_to_bf16(float f) {
  union { float f; unsigned int u; } cv; cv.f = f;
  unsigned int u = cv.u;
  u += 0x7FFFu + ((u >> 16) & 1u);          // round-to-nearest-even
  return (unsigned short)(u >> 16);
}

__device__ __forceinline__ void wait_async0() {
#if __has_builtin(__builtin_amdgcn_s_wait_asynccnt)
  __builtin_amdgcn_s_wait_asynccnt(0);
#else
  asm volatile("s_wait_asynccnt 0x0" ::: "memory");
#endif
}

// Async copy of 16 bytes global -> LDS (GLOBAL_LOAD_ASYNC_TO_LDS_B128,
// tracked by ASYNCcnt). ldsOff = low 32 bits of the flat shared pointer,
// which per the CDNA5 LDS-aperture mapping is the LDS byte address.
__device__ __forceinline__ void async_g2l_b128(unsigned ldsOff, const void* gptr) {
  asm volatile("global_load_async_to_lds_b128 %0, %1, off"
               :: "v"(ldsOff), "v"(gptr)
               : "memory");
}

#define BM 128
#define BN 64
#define BK 32
#define APAD 8          // row stride BK+APAD = 40 ushorts = 80 B (16B aligned)

// ---------------------------------------------------------------------------
// bf16 GEMM: Y[M,N] = A[M,K] * W[N,K]^T    (A,W bf16 row-major, Y f32)
// 256 threads = 8 waves; wave grid 4(M) x 2(N); each wave: 2x2 WMMA tiles.
// Double-buffered LDS tiles filled with async global->LDS B128 copies.
// ---------------------------------------------------------------------------
__global__ __launch_bounds__(256) void k_gemm_bf16(
    const unsigned short* __restrict__ A,
    const unsigned short* __restrict__ W,
    float* __restrict__ Y, int M, int N, int K)
{
  __shared__ __align__(16) unsigned short As[2][BM][BK + APAD];
  __shared__ __align__(16) unsigned short Bs[2][BN][BK + APAD];

  const int tid  = threadIdx.x;
  const int lane = tid & 31;
  const int wave = tid >> 5;
  const int grp  = lane >> 4;      // lane group 0/1 (ISA A/B/C layouts)
  const int l16  = lane & 15;
  const int wm   = wave & 3;       // 32-row M strip
  const int wn   = wave >> 2;      // 32-col N strip

  const int blockM = blockIdx.x * BM;
  const int blockN = blockIdx.y * BN;

  FragC acc[2][2];
  #pragma unroll
  for (int a = 0; a < 2; ++a)
    #pragma unroll
    for (int b = 0; b < 2; ++b)
      #pragma unroll
      for (int r = 0; r < 8; ++r) acc[a][b].f[r] = 0.0f;

  const int kTiles = K / BK;

  // Issue async copies for K-tile kt into LDS buffer buf.
  auto issue_tile = [&](int kt, int buf) {
    const int kBase = kt * BK;
    // A tile: 128x32 bf16 = 512 x 16B chunks, 2 per thread
    #pragma unroll
    for (int j = 0; j < 2; ++j) {
      int u    = tid + 256 * j;
      int row  = u >> 2;
      int part = (u & 3) << 3;          // ushort column: 0,8,16,24
      unsigned lds = (unsigned)(unsigned long long)&As[buf][row][part];
      async_g2l_b128(lds, &A[(size_t)(blockM + row) * K + kBase + part]);
    }
    // B tile: 64x32 bf16 = 256 x 16B chunks, 1 per thread
    {
      int row  = tid >> 2;
      int part = (tid & 3) << 3;
      unsigned lds = (unsigned)(unsigned long long)&Bs[buf][row][part];
      async_g2l_b128(lds, &W[(size_t)(blockN + row) * K + kBase + part]);
    }
  };

  // Prologue: stage tile 0
  issue_tile(0, 0);
  wait_async0();
  __syncthreads();

  for (int kt = 0; kt < kTiles; ++kt) {
    const int cur = kt & 1;

    // Pipeline: start copying the next tile into the other buffer
    if (kt + 1 < kTiles) issue_tile(kt + 1, cur ^ 1);
    // And nudge the tile after next toward L2 (global_prefetch_b8)
    if (kt + 2 < kTiles) {
      const unsigned short* nA =
          &A[(size_t)(blockM + (tid >> 1)) * K + (kt + 2) * BK + ((tid & 1) << 4)];
      __builtin_prefetch(nA, 0, 0);
    }

    // --- assemble fragments per ISA 16-bit A(16x32)/B(32x16) layouts ---
    FragAB af[2], bf[2];
    #pragma unroll
    for (int tm = 0; tm < 2; ++tm) {
      const int mrow = wm * 32 + tm * 16 + l16;
      #pragma unroll
      for (int j = 0; j < 8; ++j) {
        // VGPR j holds K = {2j | 16+2(j-4)} + grp*8 (pair of halves)
        const int kk = ((j < 4) ? (2 * j) : (16 + 2 * (j - 4))) + grp * 8;
        af[tm].u32[j] = *(const unsigned int*)&As[cur][mrow][kk];
      }
    }
    #pragma unroll
    for (int tn = 0; tn < 2; ++tn) {
      const int ncol = wn * 32 + tn * 16 + l16;
      #pragma unroll
      for (int j = 0; j < 8; ++j) {
        // VGPR j holds K = {2j | 16+2(j-4)} + grp*8 (pair of halves)
        bf[tn].u32[j] = *(const unsigned int*)&Bs[cur][ncol][((j < 4) ? (2 * j) : (16 + 2 * (j - 4))) + grp * 8];
      }
    }

    #pragma unroll
    for (int tm = 0; tm < 2; ++tm)
      #pragma unroll
      for (int tn = 0; tn < 2; ++tn)
        acc[tm][tn].v = __builtin_amdgcn_wmma_f32_16x16x32_bf16(
            false, af[tm].v, false, bf[tn].v, (short)0, acc[tm][tn].v,
            false, false);

    // Drain this wave's async copies, then block barrier so buffers flip safely
    wait_async0();
    __syncthreads();
  }

  // --- store: C/D layout: VGPR r -> M = r + 8*grp, N = l16 ---
  #pragma unroll
  for (int tm = 0; tm < 2; ++tm)
    #pragma unroll
    for (int tn = 0; tn < 2; ++tn)
      #pragma unroll
      for (int r = 0; r < 8; ++r) {
        const int m = blockM + wm * 32 + tm * 16 + grp * 8 + r;
        const int n = blockN + wn * 32 + tn * 16 + l16;
        Y[(size_t)m * N + n] = acc[tm][tn].f[r];
      }
}

// ---------------------------------------------------------------------------
// f32 -> bf16 cast (weights / activations)
// ---------------------------------------------------------------------------
__global__ void k_cast_bf16(const float* __restrict__ src,
                            unsigned short* __restrict__ dst, int n) {
  int t = blockIdx.x * blockDim.x + threadIdx.x;
  if (t < n) dst[t] = f32_to_bf16(src[t]);
}

// ---------------------------------------------------------------------------
// Layer 1: y1[m, c] = sum_j W1[c,3] * x[m,3]   (K=3 too small for WMMA)
// ---------------------------------------------------------------------------
__global__ void k_mlp1(const float* __restrict__ x, const float* __restrict__ W1,
                       float* __restrict__ y1, int M) {
  int t = blockIdx.x * blockDim.x + threadIdx.x;
  if (t >= M * 64) return;
  int m = t >> 6, c = t & 63;
  float s = W1[c * 3 + 0] * x[m * 3 + 0] +
            W1[c * 3 + 1] * x[m * 3 + 1] +
            W1[c * 3 + 2] * x[m * 3 + 2];
  y1[(size_t)m * 64 + c] = s;
}

// ---------------------------------------------------------------------------
// Per-channel sum / sum-of-squares over M rows (one block per channel)
// ---------------------------------------------------------------------------
__global__ __launch_bounds__(256) void k_colsum(const float* __restrict__ Y,
                                                int M, int C,
                                                float* __restrict__ sum,
                                                float* __restrict__ ssq) {
  int c = blockIdx.x;
  float s = 0.f, q = 0.f;
  for (int m = threadIdx.x; m < M; m += 256) {
    float v = Y[(size_t)m * C + c];
    s += v; q += v * v;
  }
  __shared__ float rs[256], rq[256];
  rs[threadIdx.x] = s; rq[threadIdx.x] = q;
  __syncthreads();
  for (int o = 128; o > 0; o >>= 1) {
    if (threadIdx.x < o) {
      rs[threadIdx.x] += rs[threadIdx.x + o];
      rq[threadIdx.x] += rq[threadIdx.x + o];
    }
    __syncthreads();
  }
  if (threadIdx.x == 0) { sum[c] = rs[0]; ssq[c] = rq[0]; }
}

// ---------------------------------------------------------------------------
// Training-mode BN (biased stats) + ReLU; optional f32 and/or bf16 outputs
// ---------------------------------------------------------------------------
__global__ void k_bn_relu(const float* __restrict__ Y,
                          const float* __restrict__ sum,
                          const float* __restrict__ ssq,
                          const float* __restrict__ g,
                          const float* __restrict__ b,
                          int M, int C,
                          float* __restrict__ outF,
                          unsigned short* __restrict__ outH) {
  size_t t = (size_t)blockIdx.x * blockDim.x + threadIdx.x;
  size_t total = (size_t)M * C;
  if (t >= total) return;
  int c = (int)(t % C);
  float invM = 1.0f / (float)M;
  float mean = sum[c] * invM;
  float var  = ssq[c] * invM - mean * mean;
  float v = (Y[t] - mean) * rsqrtf(var + 1e-5f) * g[c] + b[c];
  v = fmaxf(v, 0.0f);
  if (outF) outF[t] = v;
  if (outH) outH[t] = f32_to_bf16(v);
}

// ---------------------------------------------------------------------------
// Critical-point layer: one block (1024 threads) per sample.
// feats: [B, N=1024, C=1024] f32 (post BN+ReLU, so >= 0).
// ---------------------------------------------------------------------------
__global__ __launch_bounds__(1024) void k_cpl(const float* __restrict__ feats,
                                              int* __restrict__ idxOut) {
  const int b = blockIdx.x;
  const int t = threadIdx.x;
  __shared__ float scores[1024];
  __shared__ float skey[1024];
  __shared__ int   sidx[1024];
  __shared__ int   mcount;
  scores[t] = 0.0f;
  if (t == 0) mcount = 0;
  __syncthreads();

  // Per-channel first-argmax over the 1024 points (thread t = channel t)
  const float* fb = feats + (size_t)b * 1024 * 1024;
  float best = -1.0f; int bi = 0;
  for (int n = 0; n < 1024; ++n) {
    float v = fb[(size_t)n * 1024 + t];
    if (v > best) { best = v; bi = n; }   // strict '>' => first argmax
  }
  atomicAdd(&scores[bi], best);
  __syncthreads();

  if (scores[t] > 0.0f) atomicAdd(&mcount, 1);
  skey[t] = scores[t];
  sidx[t] = t;
  __syncthreads();

  // Bitonic sort: (score desc, idx asc). Since scores >= 0, this equals the
  // reference's (-score + eps*idx) stable sort with contributing packed first.
  for (int k = 2; k <= 1024; k <<= 1) {
    for (int j = k >> 1; j > 0; j >>= 1) {
      int p = t ^ j;
      if (p > t) {
        bool up = ((t & k) == 0);
        float k0 = skey[t], k1 = skey[p];
        int   i0 = sidx[t], i1 = sidx[p];
        bool pBefore = (k1 > k0) || (k1 == k0 && i1 < i0);
        if (up ? pBefore : !pBefore) {
          skey[t] = k1; skey[p] = k0;
          sidx[t] = i1; sidx[p] = i0;
        }
      }
      __syncthreads();
    }
  }

  const int m = mcount;
  if (t < 256) {
    int meff = (m > 0) ? m : 1024;
    int pos;
    if (meff >= 256) {
      pos = t;
    } else {
      float lin = (float)t * (float)(meff - 1) / 255.0f;
      pos = (int)rintf(lin);            // round-half-even like jnp.round
      if (pos < 0) pos = 0;
      if (pos > meff - 1) pos = meff - 1;
    }
    idxOut[b * 256 + t] = sidx[pos];
  }
}

// ---------------------------------------------------------------------------
// Gather critical points -> bf16 [B*K, 1024]
// ---------------------------------------------------------------------------
__global__ void k_gather_bf16(const float* __restrict__ feats,
                              const int* __restrict__ idx,
                              unsigned short* __restrict__ crit) {
  size_t t = (size_t)blockIdx.x * 256 + threadIdx.x;  // B*K*1024 total
  int c = (int)(t & 1023);
  size_t row = t >> 10;            // b*K + k
  int b = (int)(row >> 8);         // K = 256
  int point = idx[row];
  crit[t] = f32_to_bf16(feats[((size_t)b * 1024 + point) * 1024 + c]);
}

// ---------------------------------------------------------------------------
// Max-pool over K=256 critical points per (b, c)
// ---------------------------------------------------------------------------
__global__ void k_maxpool(const float* __restrict__ hc, float* __restrict__ gfeat) {
  int t = blockIdx.x * 256 + threadIdx.x;   // 64*1024
  int c = t & 1023, b = t >> 10;
  const float* p = hc + (size_t)b * 256 * 1024 + c;
  float mx = -3.4e38f;
  for (int k = 0; k < 256; ++k) mx = fmaxf(mx, p[(size_t)k * 1024]);
  gfeat[t] = mx;
}

// ---------------------------------------------------------------------------
// FC layer: out[b,o] = in[b,:] . w[o,:] + bias[o]
// ---------------------------------------------------------------------------
__global__ void k_fc(const float* __restrict__ in, const float* __restrict__ w,
                     const float* __restrict__ bias, float* __restrict__ out,
                     int B, int I, int O) {
  int t = blockIdx.x * blockDim.x + threadIdx.x;
  if (t >= B * O) return;
  int b = t / O, o = t % O;
  float s = bias[o];
  const float* ip = in + (size_t)b * I;
  const float* wp = w + (size_t)o * I;
  for (int i = 0; i < I; ++i) s += ip[i] * wp[i];
  out[t] = s;
}

// ---------------------------------------------------------------------------
// BN over batch axis only (biased) + ReLU, in place. Thread per feature.
// ---------------------------------------------------------------------------
__global__ void k_bnb_relu(float* __restrict__ y, const float* __restrict__ g,
                           const float* __restrict__ bb, int B, int O) {
  int o = blockIdx.x * blockDim.x + threadIdx.x;
  if (o >= O) return;
  float s = 0.f, q = 0.f;
  for (int b = 0; b < B; ++b) { float v = y[(size_t)b * O + o]; s += v; q += v * v; }
  float invB = 1.0f / (float)B;
  float mean = s * invB, var = q * invB - mean * mean;
  float inv = rsqrtf(var + 1e-5f);
  for (int b = 0; b < B; ++b) {
    float v = (y[(size_t)b * O + o] - mean) * inv * g[o] + bb[o];
    y[(size_t)b * O + o] = fmaxf(v, 0.0f);
  }
}

// ---------------------------------------------------------------------------
// Host launch
// ---------------------------------------------------------------------------
static inline size_t alignUp256(size_t x) { return (x + 255) & ~(size_t)255; }

extern "C" void kernel_launch(void* const* d_in, const int* in_sizes, int n_in,
                              void* d_out, int out_size, void* d_ws, size_t ws_size,
                              hipStream_t stream) {
  (void)in_sizes; (void)n_in; (void)out_size; (void)ws_size;

  const float* x     = (const float*)d_in[0];
  const float* W1    = (const float*)d_in[1];
  const float* g1    = (const float*)d_in[2];
  const float* b1    = (const float*)d_in[3];
  const float* W2    = (const float*)d_in[4];
  const float* g2    = (const float*)d_in[5];
  const float* b2    = (const float*)d_in[6];
  const float* W3    = (const float*)d_in[7];
  const float* g3    = (const float*)d_in[8];
  const float* b3    = (const float*)d_in[9];
  const float* Wc    = (const float*)d_in[10];
  const float* gc    = (const float*)d_in[11];
  const float* bc    = (const float*)d_in[12];
  const float* fc1_w = (const float*)d_in[13];
  const float* fc1_b = (const float*)d_in[14];
  const float* bn1_g = (const float*)d_in[15];
  const float* bn1_b = (const float*)d_in[16];
  const float* fc2_w = (const float*)d_in[17];
  const float* fc2_b = (const float*)d_in[18];
  const float* bn2_g = (const float*)d_in[19];
  const float* bn2_b = (const float*)d_in[20];
  const float* fc3_w = (const float*)d_in[21];
  const float* fc3_b = (const float*)d_in[22];
  float* logits = (float*)d_out;

  const int B  = 64;
  const int Np = 1024;
  const int M1 = B * Np;     // 65536
  const int Kc = 256;
  const int Mc = B * Kc;     // 16384

  // Workspace carve-out
  char* ws = (char*)d_ws;
  size_t off = 0;
  auto alloc = [&](size_t bytes) -> void* {
    void* p = ws + off;
    off = alignUp256(off + bytes);
    return p;
  };
  unsigned short* w2bf  = (unsigned short*)alloc((size_t)128 * 64 * 2);
  unsigned short* w3bf  = (unsigned short*)alloc((size_t)1024 * 128 * 2);
  unsigned short* wcbf  = (unsigned short*)alloc((size_t)1024 * 1024 * 2);
  unsigned short* h1bf  = (unsigned short*)alloc((size_t)M1 * 64 * 2);
  unsigned short* h2bf  = (unsigned short*)alloc((size_t)M1 * 128 * 2);
  unsigned short* critb = (unsigned short*)alloc((size_t)Mc * 1024 * 2);
  float* y1    = (float*)alloc((size_t)M1 * 64 * 4);
  float* y2    = (float*)alloc((size_t)M1 * 128 * 4);
  float* h3    = (float*)alloc((size_t)M1 * 1024 * 4);   // y3 / feats in place
  float* yc    = (float*)alloc((size_t)Mc * 1024 * 4);   // yc / hc in place
  float* gfeat = (float*)alloc((size_t)B * 1024 * 4);
  float* yf1   = (float*)alloc((size_t)B * 512 * 4);
  float* yf2   = (float*)alloc((size_t)B * 256 * 4);
  float* csum  = (float*)alloc((size_t)1024 * 4);
  float* cssq  = (float*)alloc((size_t)1024 * 4);
  int*   idx   = (int*)alloc((size_t)B * Kc * 4);

  // Weight casts to bf16
  k_cast_bf16<<<(128 * 64 + 255) / 256, 256, 0, stream>>>(W2, w2bf, 128 * 64);
  k_cast_bf16<<<(1024 * 128 + 255) / 256, 256, 0, stream>>>(W3, w3bf, 1024 * 128);
  k_cast_bf16<<<(1024 * 1024 + 255) / 256, 256, 0, stream>>>(Wc, wcbf, 1024 * 1024);

  // Layer 1 (K=3): scalar
  k_mlp1<<<(M1 * 64) / 256, 256, 0, stream>>>(x, W1, y1, M1);
  k_colsum<<<64, 256, 0, stream>>>(y1, M1, 64, csum, cssq);
  k_bn_relu<<<(unsigned)(((size_t)M1 * 64 + 255) / 256), 256, 0, stream>>>(
      y1, csum, cssq, g1, b1, M1, 64, nullptr, h1bf);

  // Layer 2: WMMA GEMM 65536x128x64
  { dim3 g(M1 / BM, 128 / BN);
    k_gemm_bf16<<<g, 256, 0, stream>>>(h1bf, w2bf, y2, M1, 128, 64); }
  k_colsum<<<128, 256, 0, stream>>>(y2, M1, 128, csum, cssq);
  k_bn_relu<<<(unsigned)(((size_t)M1 * 128 + 255) / 256), 256, 0, stream>>>(
      y2, csum, cssq, g2, b2, M1, 128, nullptr, h2bf);

  // Layer 3: WMMA GEMM 65536x1024x128
  { dim3 g(M1 / BM, 1024 / BN);
    k_gemm_bf16<<<g, 256, 0, stream>>>(h2bf, w3bf, h3, M1, 1024, 128); }
  k_colsum<<<1024, 256, 0, stream>>>(h3, M1, 1024, csum, cssq);
  k_bn_relu<<<(unsigned)(((size_t)M1 * 1024 + 255) / 256), 256, 0, stream>>>(
      h3, csum, cssq, g3, b3, M1, 1024, h3, nullptr);   // feats in place

  // Critical point selection + gather
  k_cpl<<<B, 1024, 0, stream>>>(h3, idx);
  k_gather_bf16<<<(unsigned)(((size_t)Mc * 1024) / 256), 256, 0, stream>>>(h3, idx, critb);

  // MLPc: WMMA GEMM 16384x1024x1024
  { dim3 g(Mc / BM, 1024 / BN);
    k_gemm_bf16<<<g, 256, 0, stream>>>(critb, wcbf, yc, Mc, 1024, 1024); }
  k_colsum<<<1024, 256, 0, stream>>>(yc, Mc, 1024, csum, cssq);
  k_bn_relu<<<(unsigned)(((size_t)Mc * 1024 + 255) / 256), 256, 0, stream>>>(
      yc, csum, cssq, gc, bc, Mc, 1024, yc, nullptr);   // hc in place

  // Global feature + FC head
  k_maxpool<<<(B * 1024) / 256, 256, 0, stream>>>(yc, gfeat);
  k_fc<<<(B * 512 + 255) / 256, 256, 0, stream>>>(gfeat, fc1_w, fc1_b, yf1, B, 1024, 512);
  k_bnb_relu<<<(512 + 255) / 256, 256, 0, stream>>>(yf1, bn1_g, bn1_b, B, 512);
  k_fc<<<(B * 256 + 255) / 256, 256, 0, stream>>>(yf1, fc2_w, fc2_b, yf2, B, 512, 256);
  k_bnb_relu<<<(256 + 255) / 256, 256, 0, stream>>>(yf2, bn2_g, bn2_b, B, 256);
  k_fc<<<(B * 40 + 255) / 256, 256, 0, stream>>>(yf2, fc3_w, fc3_b, logits, B, 256, 40);
}
